// Wasserstein_58411555226334
// MI455X (gfx1250) — matrix operand order repaired and checked by reading, not compiled, not run on hardware
//
#include <hip/hip_runtime.h>

// Problem constants (from reference): B=32, N=M=1024, REG=0.1, NIT=500
#define BB  32
#define NN  1024
#define MM  1024
#define NITER 500
#define INV_REG 10.0f   // 1/REG

typedef _Float16 h16;
typedef __attribute__((ext_vector_type(16))) _Float16 v16h;
typedef __attribute__((ext_vector_type(8)))  _Float16 v8h;
typedef __attribute__((ext_vector_type(8)))  float    v8f;

// ---------------------------------------------------------------------------
// 1) K = exp(-C/REG) in fp16, stored both row-major [b][n][m] and transposed
//    [b][m][n] so BOTH Sinkhorn GEMV directions stream contiguous rows.
//    64x64 tile per block, LDS transpose for coalesced writes.
// ---------------------------------------------------------------------------
__global__ void k_prep(const float* __restrict__ C,
                       h16* __restrict__ Kh,    // [b][n][m]
                       h16* __restrict__ KhT) { // [b][m][n]
  __shared__ h16 tile[64][65];
  const int b  = blockIdx.z;
  const int n0 = blockIdx.y * 64;
  const int m0 = blockIdx.x * 64;
  const int tx = threadIdx.x & 63;   // col in tile
  const int ty = threadIdx.x >> 6;   // 0..3
  const float* Cb  = C   + (size_t)b * NN * MM;
  h16* Khb  = Kh  + (size_t)b * NN * MM;
  h16* KhTb = KhT + (size_t)b * NN * MM;
  for (int rr = ty; rr < 64; rr += 4) {
    float c = Cb[(size_t)(n0 + rr) * MM + m0 + tx];
    h16 k = (h16)__expf(-c * INV_REG);
    Khb[(size_t)(n0 + rr) * MM + m0 + tx] = k;
    tile[rr][tx] = k;
  }
  __syncthreads();
  for (int rr = ty; rr < 64; rr += 4) {
    KhTb[(size_t)(m0 + rr) * NN + n0 + tx] = tile[tx][rr];
  }
}

// ---------------------------------------------------------------------------
// 2) init: u~ = 1 (fp16), scale_u = 1
// ---------------------------------------------------------------------------
__global__ void k_init(h16* __restrict__ uh, float* __restrict__ su) {
  int i = blockIdx.x * blockDim.x + threadIdx.x;
  if (i < BB * NN) uh[i] = (h16)1.0f;
  if (i < BB)      su[i] = 1.0f;
}

// ---------------------------------------------------------------------------
// 3) per-batch max-normalize fp32 vector into fp16 (fp16 range protection).
//    scale[b] = max(x_b); xh = x / scale. One block per batch.
// ---------------------------------------------------------------------------
__global__ void k_norm(const float* __restrict__ x, h16* __restrict__ xh,
                       float* __restrict__ scale, int L) {
  __shared__ float red[256];
  const int b = blockIdx.x;
  const float* xb = x + (size_t)b * L;
  float m = 0.0f;
  for (int i = threadIdx.x; i < L; i += 256) m = fmaxf(m, xb[i]); // x > 0
  red[threadIdx.x] = m;
  __syncthreads();
  for (int s = 128; s > 0; s >>= 1) {
    if (threadIdx.x < s) red[threadIdx.x] = fmaxf(red[threadIdx.x], red[threadIdx.x + s]);
    __syncthreads();
  }
  m = red[0];
  float inv = 1.0f / m;
  if (threadIdx.x == 0) scale[b] = m;
  for (int i = threadIdx.x; i < L; i += 256) xh[(size_t)b * L + i] = (h16)(xb[i] * inv);
}

// ---------------------------------------------------------------------------
// 4) WMMA batched GEMV:  out[b][r] = numer[b][r] / (scale[b] * sum_c K[b][r][c]*xh[b][c])
//    K row-major [b][1024][1024]. One wave -> 16 output rows.
//    A-operand = 16x32 K tile per ISA 16-bit A layout:
//      lane L: row = L%16, half = L/16; VGPR0-3 = k in [8h,8h+8),
//      VGPR4-7 = k in [16+8h,16+8h+8)  => two contiguous b128 loads per lane.
//    B-operand = xh chunk broadcast across columns: lane needs xh[k0+8h .. +8) and xh[k0+16+8h .. +8)
//      => D[i][j] identical over j = the row-i dot product (f32 accumulate).
// ---------------------------------------------------------------------------
__global__ void k_gemv16(const h16* __restrict__ Kmat, const h16* __restrict__ xh,
                         const float* __restrict__ scale, const float* __restrict__ numer,
                         float* __restrict__ out) {
  const int lane = threadIdx.x & 31;
  const int wave = threadIdx.x >> 5;                 // 0..7 (block = 256 = 8 waves)
  const int b    = blockIdx.y;
  const int row0 = blockIdx.x * 128 + wave * 16;     // 16 rows per wave
  const int half = lane >> 4;                        // 0/1
  const int r    = lane & 15;                        // A row within tile
  const h16* rowptr = Kmat + (size_t)b * NN * MM + (size_t)(row0 + r) * MM;
  const h16* xb     = xh   + (size_t)b * MM;

  v8f acc = {};
  for (int k0 = 0; k0 < 1024; k0 += 32) {
    // Prefetch the next A tile chunk for this row (global_prefetch_b8).
    if (k0 + 64 < 1024) __builtin_prefetch(rowptr + k0 + 64, 0, 3);

    v8h a_lo = *(const v8h*)(rowptr + k0 + 8 * half);
    v8h a_hi = *(const v8h*)(rowptr + k0 + 16 + 8 * half);
    v8h b_lo = *(const v8h*)(xb + k0 + 8 * half);
    v8h b_hi = *(const v8h*)(xb + k0 + 16 + 8 * half);
    v16h A  = __builtin_shufflevector(a_lo, a_hi, 0,1,2,3,4,5,6,7,8,9,10,11,12,13,14,15);
    v16h Bm = __builtin_shufflevector(b_lo, b_hi, 0,1,2,3,4,5,6,7,8,9,10,11,12,13,14,15);
    acc = __builtin_amdgcn_wmma_f32_16x16x32_f16(
        /*neg_a=*/false, A, /*neg_b=*/false, Bm,
        /*c_mod=*/(short)0, acc, /*reuse_a=*/false, /*reuse_b=*/false);
  }
  // C/D layout: VGPR i = row (i + 8*half) at col = lane&15. Col 0 -> lanes 0,16.
  if ((lane & 15) == 0) {
    const int rbase = row0 + 8 * half;
    const float s = scale[b];
    const size_t base = (size_t)b * 1024 + rbase;
    #pragma unroll
    for (int i = 0; i < 8; ++i) out[base + i] = numer[base + i] / (s * acc[i]);
  }
}

// ---------------------------------------------------------------------------
// 5) finalize: T = u*K*v (fp32 K recomputed from C for precision), copy C,
//    deterministic per-(b,n) partial of sum(C*T).
// ---------------------------------------------------------------------------
__global__ void k_final(const float* __restrict__ C, const float* __restrict__ u,
                        const float* __restrict__ v, float* __restrict__ out_C,
                        float* __restrict__ out_T, float* __restrict__ part) {
  __shared__ float red[256];
  const int b = blockIdx.y;
  const int n = blockIdx.x;
  const size_t base = ((size_t)b * NN + n) * MM;
  const float un = u[(size_t)b * NN + n];
  float p = 0.0f;
  for (int m = threadIdx.x; m < MM; m += 256) {
    float c = C[base + m];
    float t = un * __expf(-c * INV_REG) * v[(size_t)b * MM + m];
    out_C[base + m] = c;
    out_T[base + m] = t;
    p += c * t;
  }
  red[threadIdx.x] = p;
  __syncthreads();
  for (int s = 128; s > 0; s >>= 1) {
    if (threadIdx.x < s) red[threadIdx.x] += red[threadIdx.x + s];
    __syncthreads();
  }
  if (threadIdx.x == 0) part[(size_t)b * NN + n] = red[0];
}

__global__ void k_sum(const float* __restrict__ part, float* __restrict__ out_d) {
  __shared__ float red[256];
  const int b = blockIdx.x;
  float s = 0.0f;
  for (int i = threadIdx.x; i < NN; i += 256) s += part[(size_t)b * NN + i];
  red[threadIdx.x] = s;
  __syncthreads();
  for (int st = 128; st > 0; st >>= 1) {
    if (threadIdx.x < st) red[threadIdx.x] += red[threadIdx.x + st];
    __syncthreads();
  }
  if (threadIdx.x == 0) out_d[b] = red[0];
}

// ---------------------------------------------------------------------------
extern "C" void kernel_launch(void* const* d_in, const int* in_sizes, int n_in,
                              void* d_out, int out_size, void* d_ws, size_t ws_size,
                              hipStream_t stream) {
  const float* C  = (const float*)d_in[0];   // [B,N,M]
  const float* d1 = (const float*)d_in[1];   // [B,N]
  const float* d2 = (const float*)d_in[2];   // [B,M]

  // d_out = [dists(32) | C(32M) | T(32M)] fp32.
  float* out_d = (float*)d_out;
  float* out_C = out_d + BB;
  float* out_T = out_C + (size_t)BB * NN * MM;

  // Stash the two fp16 K layouts inside the (not-yet-needed) 128MB output
  // regions: each needs only 64MB. Both fit in the 192MB L2 together.
  h16* Kh  = (h16*)out_C;   // row-major  [b][n][m]
  h16* KhT = (h16*)out_T;   // transposed [b][m][n]

  // Small fp32/fp16 vectors in workspace (~0.6 MB).
  char* ws = (char*)d_ws;
  size_t off = 0;
  float* u    = (float*)(ws + off); off += (size_t)BB * NN * 4;
  float* v    = (float*)(ws + off); off += (size_t)BB * MM * 4;
  h16*   uh   = (h16*)  (ws + off); off += (size_t)BB * NN * 2;
  h16*   vh   = (h16*)  (ws + off); off += (size_t)BB * MM * 2;
  float* su   = (float*)(ws + off); off += 64 * 4;
  float* tv   = (float*)(ws + off); off += 64 * 4;
  float* part = (float*)(ws + off); off += (size_t)BB * NN * 4;

  dim3 gp(MM / 64, NN / 64, BB);
  k_prep<<<gp, 256, 0, stream>>>(C, Kh, KhT);
  k_init<<<(BB * NN + 255) / 256, 256, 0, stream>>>(uh, su);

  dim3 gg(NN / 128, BB);   // 8 x 32 blocks of 8 waves; 16 rows/wave
  for (int it = 0; it < NITER; ++it) {
    // v = d2 / (K^T u):   K^T rows are KhT rows (contiguous).
    k_gemv16<<<gg, 256, 0, stream>>>(KhT, uh, su, d2, v);
    k_norm<<<BB, 256, 0, stream>>>(v, vh, tv, MM);
    // u = d1 / (K v):     Kh rows contiguous.
    k_gemv16<<<gg, 256, 0, stream>>>(Kh, vh, tv, d1, u);
    k_norm<<<BB, 256, 0, stream>>>(u, uh, su, NN);
  }

  // Finalize overwrites the stashed Kh/KhT regions with C and T.
  dim3 gf(NN, BB);
  k_final<<<gf, 256, 0, stream>>>(C, u, v, out_C, out_T, part);
  k_sum<<<BB, 256, 0, stream>>>(part, out_d);
}
